// TopKGating_47038481825968
// MI455X (gfx1250) — matrix operand order repaired and checked by reading, not compiled, not run on hardware
//
#include <hip/hip_runtime.h>
#include <hip/hip_bf16.h>
#include <math.h>

// CDNA5 / gfx1250, wave32. bf16 WMMA with f32 accumulate.
typedef __attribute__((ext_vector_type(16))) __bf16 v16bf;
typedef __attribute__((ext_vector_type(8)))  __bf16 v8bf;
typedef __attribute__((ext_vector_type(8)))  float  v8f;
typedef __attribute__((ext_vector_type(4)))  float  v4f;   // clang vector: OK for NT builtins

#define TOKENS   16384
#define DIM      4096
#define EXPERTS  64
#define TOPK     8
#define KTILE    32          // K per WMMA step
#define ROWS_PB  64          // rows per block (4 waves x 16 rows)
#define PITCH    40          // bf16 elements per LDS row: 80B, 16B-aligned, conflict-free

// ---------------------------------------------------------------------------
// Kernel 1: 64x64 logits tile via v_wmma_f32_16x16x32_bf16, then per-row
// softmax -> p, unbiased std, per-block std partial sum.
// bf16 tiles live in LDS; fragments are contiguous 16B runs -> ds_load_b128.
// ---------------------------------------------------------------------------
__global__ __launch_bounds__(128) void gate_gemm_softmax_kernel(
    const float* __restrict__ x,        // [TOKENS, DIM]
    const float* __restrict__ gw,       // [EXPERTS, DIM]
    const float* __restrict__ gb,       // [EXPERTS]
    float* __restrict__ p_out,          // [TOKENS, EXPERTS]
    float* __restrict__ std_part)       // [gridDim.x]
{
    __shared__ __bf16 xsb[ROWS_PB * PITCH];   // 64 rows x 32 K (bf16)
    __shared__ __bf16 wsb[EXPERTS * PITCH];   // 64 experts x 32 K (bf16)
    __shared__ float  lg [ROWS_PB * 65];      // 64 x 64 logits tile (pad 65)
    __shared__ float  stds[ROWS_PB];

    const int tid  = threadIdx.x;
    const int wave = tid >> 5;            // 0..3, owns rows 16*wave..+15
    const int lane = tid & 31;
    const int r0   = blockIdx.x * ROWS_PB;

    v8f acc[4];
    #pragma unroll
    for (int t = 0; t < 4; ++t) acc[t] = (v8f){0.f,0.f,0.f,0.f,0.f,0.f,0.f,0.f};

    const int lrow  = tid >> 1;           // 0..63: LDS staging row
    const int lhalf = (tid & 1) * 16;     // 0 or 16: K half staged by this thread

    const int m     = 16 * wave + (lane & 15);  // A-fragment row (M)
    const int khalf = lane >> 4;                // A: K 0-7/16-23 vs 8-15/24-31

    for (int k0 = 0; k0 < DIM; k0 += KTILE) {
        // ---- stage bf16 tiles: load f32, convert once, store 16B chunks ----
        const v4f* xsrc = (const v4f*)(x  + (size_t)(r0 + lrow) * DIM + k0 + lhalf);
        const v4f* wsrc = (const v4f*)(gw + (size_t)lrow        * DIM + k0 + lhalf);

        v4f x0 = __builtin_nontemporal_load(xsrc + 0);   // x streamed once: NT
        v4f x1 = __builtin_nontemporal_load(xsrc + 1);
        v4f x2 = __builtin_nontemporal_load(xsrc + 2);
        v4f x3 = __builtin_nontemporal_load(xsrc + 3);
        v4f w0 = wsrc[0], w1 = wsrc[1], w2 = wsrc[2], w3 = wsrc[3];

        v8bf xa, xb, wa, wb;
        #pragma unroll
        for (int j = 0; j < 4; ++j) {
            xa[j]     = (__bf16)x0[j];
            xa[j + 4] = (__bf16)x1[j];
            xb[j]     = (__bf16)x2[j];
            xb[j + 4] = (__bf16)x3[j];
            wa[j]     = (__bf16)w0[j];
            wa[j + 4] = (__bf16)w1[j];
            wb[j]     = (__bf16)w2[j];
            wb[j + 4] = (__bf16)w3[j];
        }

        *(v8bf*)&xsb[lrow*PITCH + lhalf]     = xa;   // 16B aligned stores
        *(v8bf*)&xsb[lrow*PITCH + lhalf + 8] = xb;
        *(v8bf*)&wsb[lrow*PITCH + lhalf]     = wa;
        *(v8bf*)&wsb[lrow*PITCH + lhalf + 8] = wb;
        __syncthreads();

        // ---- A fragment: two contiguous 16B runs (ds_load_b128 x2) ----
        v8bf alo = *(const v8bf*)&xsb[m*PITCH +      khalf*8];  // K = khalf*8 + 0..7
        v8bf ahi = *(const v8bf*)&xsb[m*PITCH + 16 + khalf*8];  // K = 16+khalf*8 + 0..7
        v16bf a = __builtin_shufflevector(alo, ahi,
                     0,1,2,3,4,5,6,7,8,9,10,11,12,13,14,15);

        // ---- 4 N-tiles: B fragment = two contiguous 16B runs (ds_load_b128 x2) ----
        #pragma unroll
        for (int t = 0; t < 4; ++t) {
            const int n = 16 * t + (lane & 15);                 // B column (expert)
            v8bf blo = *(const v8bf*)&wsb[n*PITCH + khalf*8];       // K = khalf*8+0..7
            v8bf bhi = *(const v8bf*)&wsb[n*PITCH + 16 + khalf*8];  // K = 16+khalf*8+0..7
            v16bf b = __builtin_shufflevector(blo, bhi,
                         0,1,2,3,4,5,6,7,8,9,10,11,12,13,14,15);
            acc[t] = __builtin_amdgcn_wmma_f32_16x16x32_bf16(
                         /*neg_a=*/false, a, /*neg_b=*/false, b,
                         /*c_mod=*/(short)0, acc[t],
                         /*reuse_a=*/false, /*reuse_b=*/false);
        }
        __syncthreads();
    }

    // ---- bias add + spill fragments to LDS logits tile ----
    // C/D layout: vgpr r, lane l -> M = 16*wave + r + 8*(l>>4), N = 16*t + (l&15)
    #pragma unroll
    for (int t = 0; t < 4; ++t) {
        const int n  = 16 * t + (lane & 15);
        const float bv = gb[n];
        #pragma unroll
        for (int r = 0; r < 8; ++r) {
            const int mm = 16 * wave + r + 8 * (lane >> 4);
            lg[mm*65 + n] = acc[t][r] + bv;
        }
    }
    __syncthreads();

    // ---- per-row softmax p + unbiased std (one thread per row) ----
    if (tid < ROWS_PB) {
        float mx = -INFINITY;
        for (int i = 0; i < EXPERTS; ++i) mx = fmaxf(mx, lg[tid*65 + i]);
        float s = 0.f;
        for (int i = 0; i < EXPERTS; ++i) {
            float e = expf(lg[tid*65 + i] - mx);
            lg[tid*65 + i] = e;
            s += e;
        }
        const float inv = 1.f / s;
        float var = 0.f;
        for (int i = 0; i < EXPERTS; ++i) {
            float p = lg[tid*65 + i] * inv;
            lg[tid*65 + i] = p;
            float d = p - (1.f / 64.f);     // softmax row sums to 1 -> mean = 1/64
            var += d * d;
        }
        stds[tid] = sqrtf(var * (1.f / 63.f));  // ddof = 1
    }
    __syncthreads();

    // ---- coalesced store of p tile ----
    for (int e = tid; e < ROWS_PB * EXPERTS; e += 128) {
        const int row = e >> 6, col = e & 63;
        p_out[(size_t)r0 * EXPERTS + e] = lg[row*65 + col];
    }

    // ---- deterministic per-block std partial ----
    if (tid == 0) {
        float s = 0.f;
        for (int i = 0; i < ROWS_PB; ++i) s += stds[i];
        std_part[blockIdx.x] = s;
    }
}

// ---------------------------------------------------------------------------
// Kernel 2: deterministic serial reduction of per-block std partials.
// ---------------------------------------------------------------------------
__global__ void reduce_partials(const float* __restrict__ part,
                                float* __restrict__ total, int n)
{
    if (threadIdx.x == 0 && blockIdx.x == 0) {
        float s = 0.f;
        for (int i = 0; i < n; ++i) s += part[i];
        *total = s;
    }
}

// ---------------------------------------------------------------------------
// Kernel 3: per-row top-8 threshold on p (softmax is monotone, so the mask on
// logits equals the mask on p), alpha, log1p/expm1 blend, final softmax.
// ---------------------------------------------------------------------------
__global__ __launch_bounds__(64) void gate_finalize_kernel(
    const float* __restrict__ p_in,     // [TOKENS, EXPERTS]
    const float* __restrict__ sum_std,  // scalar: sum of all row stds
    float* __restrict__ out)            // [TOKENS, EXPERTS]
{
    __shared__ float L[ROWS_PB * 65];
    const int t = threadIdx.x;          // row within tile
    const size_t base = (size_t)blockIdx.x * ROWS_PB * EXPERTS;

    // coalesced load of the 64x64 p tile
    for (int i = 0; i < ROWS_PB; ++i)
        L[i*65 + t] = p_in[base + (size_t)i * EXPERTS + t];
    __syncthreads();

    // row t: recompute std + find 8th largest via insertion
    float top[TOPK];
    #pragma unroll
    for (int j = 0; j < TOPK; ++j) top[j] = -INFINITY;
    float var = 0.f;
    for (int i = 0; i < EXPERTS; ++i) {
        const float v = L[t*65 + i];
        const float d = v - (1.f / 64.f);
        var += d * d;
        if (v > top[TOPK - 1]) {
            int j = TOPK - 1;
            while (j > 0 && top[j - 1] < v) { top[j] = top[j - 1]; --j; }
            top[j] = v;
        }
    }
    const float stdv  = sqrtf(var * (1.f / 63.f));
    const float alpha = (*sum_std * (1.f / (float)TOKENS)) / (stdv + 1e-6f);
    const float kth   = top[TOPK - 1];

    // out = mask ? alpha*log1p(p) : alpha*expm1(p); then softmax(out)
    float mx = -INFINITY;
    for (int i = 0; i < EXPERTS; ++i) {
        const float p = L[t*65 + i];
        const float o = (p < kth) ? alpha * log1pf(p) : alpha * expm1f(p);
        L[t*65 + i] = o;
        mx = fmaxf(mx, o);
    }
    float s = 0.f;
    for (int i = 0; i < EXPERTS; ++i) {
        const float e = expf(L[t*65 + i] - mx);
        L[t*65 + i] = e;
        s += e;
    }
    const float inv = 1.f / s;
    for (int i = 0; i < EXPERTS; ++i) L[t*65 + i] *= inv;
    __syncthreads();

    // coalesced store
    for (int i = 0; i < ROWS_PB; ++i)
        out[base + (size_t)i * EXPERTS + t] = L[i*65 + t];
}

// ---------------------------------------------------------------------------
extern "C" void kernel_launch(void* const* d_in, const int* in_sizes, int n_in,
                              void* d_out, int out_size, void* d_ws, size_t ws_size,
                              hipStream_t stream)
{
    (void)in_sizes; (void)n_in; (void)out_size; (void)ws_size;

    const float* x  = (const float*)d_in[0];   // [16384, 4096]
    const float* gw = (const float*)d_in[1];   // [64, 4096]
    const float* gb = (const float*)d_in[2];   // [64]
    float*       o  = (float*)d_out;           // [16384, 64]

    // workspace layout: p (4 MB) | 256 partials | 1 total
    float* p_ws  = (float*)d_ws;
    float* part  = (float*)((char*)d_ws + (size_t)TOKENS * EXPERTS * sizeof(float));
    float* total = part + 256;

    const int nblk = TOKENS / ROWS_PB;  // 256

    gate_gemm_softmax_kernel<<<nblk, 128, 0, stream>>>(x, gw, gb, p_ws, part);
    reduce_partials<<<1, 32, 0, stream>>>(part, total, nblk);
    gate_finalize_kernel<<<nblk, 64, 0, stream>>>(p_ws, total, o);
}
